// DecoderBlock_32658931319157
// MI455X (gfx1250) — compile-verified, matrix operand order checked by reading, not run on hardware
//
#include <hip/hip_runtime.h>
#include <hip/hip_bf16.h>
#include <math.h>

// ---------------------------------------------------------------------------
// Dimensions (fixed by the reference)
// ---------------------------------------------------------------------------
#define BB 4
#define TT 1024
#define SS 1024
#define EE 1024
#define HH 16
#define DD 64
#define BT (BB * TT)      // 4096 rows of x
#define E4 (4 * EE)       // 4096 ffn hidden

typedef __attribute__((ext_vector_type(16))) __bf16 bf16x16;
typedef __attribute__((ext_vector_type(8)))  float  f32x8;

// ---------------------------------------------------------------------------
// WMMA helper (wave32, 16x16x32 bf16 -> f32)
// ---------------------------------------------------------------------------
__device__ __forceinline__ f32x8 wmma_bf16(bf16x16 a, bf16x16 b, f32x8 c) {
    return __builtin_amdgcn_wmma_f32_16x16x32_bf16(false, a, false, b,
                                                   (short)0, c, false, false);
}

// A fragment: 16x32 (MxK) row-major, stride ld. Two contiguous 8-elem groups
// per lane -> global_load_b128 pairs.
__device__ __forceinline__ bf16x16 load_a_frag(const __bf16* __restrict__ src, int ld) {
    int lane = threadIdx.x & 31;
    const __bf16* p = src + (size_t)(lane & 15) * ld + ((lane >> 4) * 8);
    bf16x16 f;
#pragma unroll
    for (int j = 0; j < 8; ++j) { f[j] = p[j]; f[j + 8] = p[16 + j]; }
    return f;
}

// B fragment from TRANSPOSED operand Bt[N,K] (row-major, stride ldT).
__device__ __forceinline__ bf16x16 load_bT_frag(const __bf16* __restrict__ srcT, int ldT) {
    int lane = threadIdx.x & 31;
    const __bf16* p = srcT + (size_t)(lane & 15) * ldT + ((lane >> 4) * 8);
    bf16x16 f;
#pragma unroll
    for (int j = 0; j < 8; ++j) { f[j] = p[j]; f[j + 8] = p[16 + j]; }
    return f;
}

// ---------------------------------------------------------------------------
// fp32 -> bf16 conversion (flat)
// ---------------------------------------------------------------------------
__global__ void cvt_f32_bf16_kernel(const float* __restrict__ in,
                                    __bf16* __restrict__ out, int n) {
    int i = blockIdx.x * blockDim.x + threadIdx.x;
    if (i < n) out[i] = (__bf16)in[i];
}

// fp32 [batch, R, C] -> bf16 transposed [batch, C, R]  (one-time weight prep)
__global__ void cvtT_f32_bf16_kernel(const float* __restrict__ in,
                                     __bf16* __restrict__ out, int R, int C) {
    int b = blockIdx.y;
    int i = blockIdx.x * blockDim.x + threadIdx.x;
    int n = R * C;
    if (i >= n) return;
    int r = i / C, c = i - r * C;
    out[((size_t)b * C + c) * R + r] = (__bf16)in[((size_t)b * R + r) * C + c];
}

// ---------------------------------------------------------------------------
// LayerNorm (fp32 in -> bf16 out), one block per row, E = 1024.
// float4 per thread: single global read, values stay in registers.
// ---------------------------------------------------------------------------
__global__ void layernorm_kernel(const float* __restrict__ x,
                                 const float* __restrict__ g,
                                 const float* __restrict__ b,
                                 __bf16* __restrict__ out) {
    __shared__ float red[256];
    int row = blockIdx.x;
    int i4  = threadIdx.x * 4;
    const float* xr = x + (size_t)row * EE;

    float4 v = *(const float4*)(xr + i4);
    float s = v.x + v.y + v.z + v.w;
    red[threadIdx.x] = s; __syncthreads();
    for (int st = 128; st > 0; st >>= 1) {
        if (threadIdx.x < st) red[threadIdx.x] += red[threadIdx.x + st];
        __syncthreads();
    }
    float mean = red[0] * (1.f / EE);
    __syncthreads();

    float dx = v.x - mean, dy = v.y - mean, dz = v.z - mean, dw = v.w - mean;
    red[threadIdx.x] = dx * dx + dy * dy + dz * dz + dw * dw;
    __syncthreads();
    for (int st = 128; st > 0; st >>= 1) {
        if (threadIdx.x < st) red[threadIdx.x] += red[threadIdx.x + st];
        __syncthreads();
    }
    float inv = rsqrtf(red[0] * (1.f / EE) + 1e-5f);

    float4 gv = *(const float4*)(g + i4);
    float4 bv = *(const float4*)(b + i4);
    union { __bf16 h[4]; uint2 u; } pk;
    pk.h[0] = (__bf16)(dx * inv * gv.x + bv.x);
    pk.h[1] = (__bf16)(dy * inv * gv.y + bv.y);
    pk.h[2] = (__bf16)(dz * inv * gv.z + bv.z);
    pk.h[3] = (__bf16)(dw * inv * gv.w + bv.w);
    *(uint2*)(out + (size_t)row * EE + i4) = pk.u;
}

// ---------------------------------------------------------------------------
// Per-head projection with PRE-TRANSPOSED weights. Double-buffered K loop.
// A: [B*1024, K] bf16 ; Wt: [H, 64, K] bf16 (d-major)
// out normal:     [B, H, 1024, 64]   (Q, K)
// out transposed: [B, H, 64, 1024]   (V -> feeds V^T fragments directly)
// ---------------------------------------------------------------------------
#define HP_RM_LD 72          // row-major stage stride (bf16), 16B multiple
#define HP_TR_LD 40          // transposed stage stride (bf16), 16B multiple
__global__ void headproj_kernel(const __bf16* __restrict__ A,
                                const __bf16* __restrict__ Wt,
                                __bf16* __restrict__ out,
                                int K, int transposeOut) {
    __shared__ __bf16 hstage[8][64 * HP_TR_LD];
    int h    = blockIdx.y;
    int lane = threadIdx.x & 31;
    int wave = threadIdx.x >> 5;
    int row0 = blockIdx.x * 256 + wave * 32;
    const __bf16* WhT = Wt + (size_t)h * DD * K;
    const __bf16* Ar0 = A + (size_t)row0 * K;
    const __bf16* Ar1 = A + (size_t)(row0 + 16) * K;

    f32x8 c[2][4] = {};
    bf16x16 a0[2], a1[2], bb[2][4];

    // preload stage 0 (k = 0)
    a0[0] = load_a_frag(Ar0, K);
    a1[0] = load_a_frag(Ar1, K);
#pragma unroll
    for (int ni = 0; ni < 4; ++ni)
        bb[0][ni] = load_bT_frag(WhT + (size_t)(ni * 16) * K, K);

    int k = 0;
    for (; k + 64 < K; k += 64) {
        // load stage 1 (k+32)
        a0[1] = load_a_frag(Ar0 + k + 32, K);
        a1[1] = load_a_frag(Ar1 + k + 32, K);
#pragma unroll
        for (int ni = 0; ni < 4; ++ni)
            bb[1][ni] = load_bT_frag(WhT + (size_t)(ni * 16) * K + k + 32, K);
        // compute stage 0 (k)
#pragma unroll
        for (int ni = 0; ni < 4; ++ni) {
            c[0][ni] = wmma_bf16(a0[0], bb[0][ni], c[0][ni]);
            c[1][ni] = wmma_bf16(a1[0], bb[0][ni], c[1][ni]);
        }
        // load stage 0 (k+64)
        a0[0] = load_a_frag(Ar0 + k + 64, K);
        a1[0] = load_a_frag(Ar1 + k + 64, K);
#pragma unroll
        for (int ni = 0; ni < 4; ++ni)
            bb[0][ni] = load_bT_frag(WhT + (size_t)(ni * 16) * K + k + 64, K);
        // compute stage 1 (k+32)
#pragma unroll
        for (int ni = 0; ni < 4; ++ni) {
            c[0][ni] = wmma_bf16(a0[1], bb[1][ni], c[0][ni]);
            c[1][ni] = wmma_bf16(a1[1], bb[1][ni], c[1][ni]);
        }
    }
    // tail: k, k+32 remain
    a0[1] = load_a_frag(Ar0 + k + 32, K);
    a1[1] = load_a_frag(Ar1 + k + 32, K);
#pragma unroll
    for (int ni = 0; ni < 4; ++ni)
        bb[1][ni] = load_bT_frag(WhT + (size_t)(ni * 16) * K + k + 32, K);
#pragma unroll
    for (int ni = 0; ni < 4; ++ni) {
        c[0][ni] = wmma_bf16(a0[0], bb[0][ni], c[0][ni]);
        c[1][ni] = wmma_bf16(a1[0], bb[0][ni], c[1][ni]);
    }
#pragma unroll
    for (int ni = 0; ni < 4; ++ni) {
        c[0][ni] = wmma_bf16(a0[1], bb[1][ni], c[0][ni]);
        c[1][ni] = wmma_bf16(a1[1], bb[1][ni], c[1][ni]);
    }

    int col0 = lane & 15, rhi = (lane >> 4) * 8;
    __bf16* sw = hstage[wave];
    int bb0 = row0 >> 10;            // rows-per-batch is always 1024
    int t0  = row0 & 1023;

    if (!transposeOut) {
#pragma unroll
        for (int mi = 0; mi < 2; ++mi)
#pragma unroll
            for (int ni = 0; ni < 4; ++ni)
#pragma unroll
                for (int r = 0; r < 8; ++r)
                    sw[(mi * 16 + r + rhi) * HP_RM_LD + ni * 16 + col0] =
                        (__bf16)c[mi][ni][r];
        int t = t0 + lane;
        __bf16* op = out + (((size_t)bb0 * HH + h) * 1024 + t) * DD;
#pragma unroll
        for (int c0 = 0; c0 < 64; c0 += 8)
            *(uint4*)(op + c0) = *(uint4*)(&sw[lane * HP_RM_LD + c0]);
    } else {
#pragma unroll
        for (int mi = 0; mi < 2; ++mi)
#pragma unroll
            for (int ni = 0; ni < 4; ++ni)
#pragma unroll
                for (int r = 0; r < 8; ++r)
                    sw[(ni * 16 + col0) * HP_TR_LD + mi * 16 + r + rhi] =
                        (__bf16)c[mi][ni][r];
#pragma unroll
        for (int dd = 0; dd < 2; ++dd) {
            int d = lane + dd * 32;
            __bf16* op = out + (((size_t)bb0 * HH + h) * DD + d) * 1024 + t0;
#pragma unroll
            for (int c0 = 0; c0 < 32; c0 += 8)
                *(uint4*)(op + c0) = *(uint4*)(&sw[d * HP_TR_LD + c0]);
        }
    }
}

// ---------------------------------------------------------------------------
// Flash attention, one wave per (b, h, 16-query tile).
// S^T = K * Q^T  (softmax over s is per-lane + one xor16 shuffle; the C
// layout of S^T coincides with the B-fragment layout of P^T).
// Q,K: [B,H,len,64] bf16 ; Vt: [B,H,64,len] bf16 ; O: [B,Tq,H*64] bf16
// ---------------------------------------------------------------------------
__global__ void attn_kernel(const __bf16* __restrict__ Q,
                            const __bf16* __restrict__ Km,
                            const __bf16* __restrict__ Vt,
                            __bf16* __restrict__ O,
                            int Tq, int Skv, int causal, float scale) {
    int b = blockIdx.z, h = blockIdx.y;
    int t_base = blockIdx.x * 16;
    int lane = threadIdx.x & 31;
    int lo = lane & 15, hi = lane >> 4;
    int kb = hi * 8;

    const __bf16* Qp = Q  + (((size_t)b * HH + h) * Tq) * DD;
    const __bf16* Kp = Km + (((size_t)b * HH + h) * Skv) * DD;
    const __bf16* Vp = Vt + (((size_t)b * HH + h) * DD) * Skv;

    bf16x16 bq[2];
    {
        const __bf16* qr = Qp + (size_t)(t_base + lo) * DD;
#pragma unroll
        for (int f = 0; f < 2; ++f)
#pragma unroll
            for (int j = 0; j < 8; ++j) {
                bq[f][j]     = qr[f * 32 + kb + j];
                bq[f][j + 8] = qr[f * 32 + kb + 16 + j];
            }
    }

    f32x8 o[4] = {};
    float m_run = -1e30f, l_run = 0.f;
    int t_glob = t_base + lo;
    int s_end = causal ? (t_base + 16) : Skv;

    for (int s0 = 0; s0 < s_end; s0 += 32) {
        f32x8 st[2];
#pragma unroll
        for (int half = 0; half < 2; ++half) {
            const __bf16* kr = Kp + (size_t)(s0 + half * 16 + lo) * DD;
            bf16x16 a0, a1;
#pragma unroll
            for (int j = 0; j < 8; ++j) {
                a0[j]     = kr[kb + j];        a0[j + 8] = kr[kb + 16 + j];
                a1[j]     = kr[32 + kb + j];   a1[j + 8] = kr[32 + kb + 16 + j];
            }
            f32x8 z = {};
            z = wmma_bf16(a0, bq[0], z);
            z = wmma_bf16(a1, bq[1], z);
            st[half] = z;
        }
        float cm = -1e30f;
#pragma unroll
        for (int half = 0; half < 2; ++half)
#pragma unroll
            for (int r = 0; r < 8; ++r) {
                int s = s0 + half * 16 + r + kb;
                float v = st[half][r] * scale;
                if (causal && s > t_glob) v = -1e9f;
                st[half][r] = v;
                cm = fmaxf(cm, v);
            }
        cm = fmaxf(cm, __shfl_xor(cm, 16, 32));
        float m_new = fmaxf(m_run, cm);
        float alpha = __expf(m_run - m_new);

        float cs = 0.f;
        bf16x16 pb;
#pragma unroll
        for (int r = 0; r < 8; ++r) {
            float p0 = __expf(st[0][r] - m_new);
            float p1 = __expf(st[1][r] - m_new);
            cs += p0 + p1;
            pb[r]     = (__bf16)p0;
            pb[r + 8] = (__bf16)p1;
        }
        cs += __shfl_xor(cs, 16, 32);
        l_run = l_run * alpha + cs;
        m_run = m_new;

#pragma unroll
        for (int dt = 0; dt < 4; ++dt) {
#pragma unroll
            for (int r = 0; r < 8; ++r) o[dt][r] *= alpha;
            bf16x16 av;
            const __bf16* vr = Vp + (size_t)(dt * 16 + lo) * Skv + s0 + kb;
#pragma unroll
            for (int j = 0; j < 8; ++j) {
                av[j]     = vr[j];
                av[j + 8] = vr[16 + j];
            }
            o[dt] = wmma_bf16(av, pb, o[dt]);
        }
    }

    float inv = 1.f / l_run;
    int t = t_base + lo;
#pragma unroll
    for (int dt = 0; dt < 4; ++dt) {
        union { __bf16 hh[8]; uint4 v; } u;
#pragma unroll
        for (int r = 0; r < 8; ++r) u.hh[r] = (__bf16)(o[dt][r] * inv);
        size_t base = (((size_t)b * Tq + t) * HH + h) * DD + dt * 16 + kb;
        *reinterpret_cast<uint4*>(O + base) = u.v;
    }
}

// ---------------------------------------------------------------------------
// Generic WMMA GEMM with TRANSPOSED B operand, double-buffered K loop:
//   C[M,N] = A[M,K](bf16) * Bt[N,K]^T (bf16) (+bias)(+resid)(ReLU?)
// Block = 256 threads = 8 waves, block tile 128x128, wave tile 32x64.
// ---------------------------------------------------------------------------
#define GM_LD 68             // f32 stage stride, 16B multiple, !=64 banks
__global__ void gemm_bf16_kernel(const __bf16* __restrict__ A,
                                 const __bf16* __restrict__ Bt,
                                 const float* __restrict__ bias,
                                 const float* __restrict__ resid,
                                 float* __restrict__ outF,
                                 __bf16* __restrict__ outB,
                                 int M, int N, int K, int relu) {
    __shared__ float cstage[8][32 * GM_LD];
    int lane  = threadIdx.x & 31;
    int wave  = threadIdx.x >> 5;
    int waveM = wave & 3, waveN = wave >> 2;
    int m0 = blockIdx.x * 128 + waveM * 32;
    int n0 = blockIdx.y * 128 + waveN * 64;
    const __bf16* Ar0 = A + (size_t)m0 * K;
    const __bf16* Ar1 = A + (size_t)(m0 + 16) * K;

    f32x8 c[2][4] = {};
    bf16x16 a0[2], a1[2], bb[2][4];

    // preload stage 0 (k = 0)
    a0[0] = load_a_frag(Ar0, K);
    a1[0] = load_a_frag(Ar1, K);
#pragma unroll
    for (int ni = 0; ni < 4; ++ni)
        bb[0][ni] = load_bT_frag(Bt + (size_t)(n0 + ni * 16) * K, K);

    int k = 0;
    for (; k + 64 < K; k += 64) {
        a0[1] = load_a_frag(Ar0 + k + 32, K);
        a1[1] = load_a_frag(Ar1 + k + 32, K);
#pragma unroll
        for (int ni = 0; ni < 4; ++ni)
            bb[1][ni] = load_bT_frag(Bt + (size_t)(n0 + ni * 16) * K + k + 32, K);
#pragma unroll
        for (int ni = 0; ni < 4; ++ni) {
            c[0][ni] = wmma_bf16(a0[0], bb[0][ni], c[0][ni]);
            c[1][ni] = wmma_bf16(a1[0], bb[0][ni], c[1][ni]);
        }
        a0[0] = load_a_frag(Ar0 + k + 64, K);
        a1[0] = load_a_frag(Ar1 + k + 64, K);
#pragma unroll
        for (int ni = 0; ni < 4; ++ni)
            bb[0][ni] = load_bT_frag(Bt + (size_t)(n0 + ni * 16) * K + k + 64, K);
#pragma unroll
        for (int ni = 0; ni < 4; ++ni) {
            c[0][ni] = wmma_bf16(a0[1], bb[1][ni], c[0][ni]);
            c[1][ni] = wmma_bf16(a1[1], bb[1][ni], c[1][ni]);
        }
    }
    // tail: k, k+32
    a0[1] = load_a_frag(Ar0 + k + 32, K);
    a1[1] = load_a_frag(Ar1 + k + 32, K);
#pragma unroll
    for (int ni = 0; ni < 4; ++ni)
        bb[1][ni] = load_bT_frag(Bt + (size_t)(n0 + ni * 16) * K + k + 32, K);
#pragma unroll
    for (int ni = 0; ni < 4; ++ni) {
        c[0][ni] = wmma_bf16(a0[0], bb[0][ni], c[0][ni]);
        c[1][ni] = wmma_bf16(a1[0], bb[0][ni], c[1][ni]);
    }
#pragma unroll
    for (int ni = 0; ni < 4; ++ni) {
        c[0][ni] = wmma_bf16(a0[1], bb[1][ni], c[0][ni]);
        c[1][ni] = wmma_bf16(a1[1], bb[1][ni], c[1][ni]);
    }

    // stage C tile [32 rows][64 cols] into this wave's LDS slice
    int col0 = lane & 15, rhi = (lane >> 4) * 8;
    float* sw = cstage[wave];
#pragma unroll
    for (int mi = 0; mi < 2; ++mi)
#pragma unroll
        for (int ni = 0; ni < 4; ++ni)
#pragma unroll
            for (int r = 0; r < 8; ++r)
                sw[(mi * 16 + r + rhi) * GM_LD + ni * 16 + col0] = c[mi][ni][r];

    // one lane per row: vectorized bias/resid/output
    int row = m0 + lane;
    const float* rp = resid ? resid + (size_t)row * N + n0 : nullptr;
    float*       fp = outF  ? outF  + (size_t)row * N + n0 : nullptr;
    __bf16*      bp = outB  ? outB  + (size_t)row * N + n0 : nullptr;
#pragma unroll
    for (int c0 = 0; c0 < 64; c0 += 4) {
        float4 vv = *(float4*)(&sw[lane * GM_LD + c0]);
        if (bias) {
            float4 b4 = *(const float4*)(bias + n0 + c0);
            vv.x += b4.x; vv.y += b4.y; vv.z += b4.z; vv.w += b4.w;
        }
        if (rp) {
            float4 r4 = *(const float4*)(rp + c0);
            vv.x += r4.x; vv.y += r4.y; vv.z += r4.z; vv.w += r4.w;
        }
        if (relu) {
            vv.x = fmaxf(vv.x, 0.f); vv.y = fmaxf(vv.y, 0.f);
            vv.z = fmaxf(vv.z, 0.f); vv.w = fmaxf(vv.w, 0.f);
        }
        if (fp) *(float4*)(fp + c0) = vv;
        if (bp) {
            union { __bf16 h[4]; uint2 u; } pk;
            pk.h[0] = (__bf16)vv.x; pk.h[1] = (__bf16)vv.y;
            pk.h[2] = (__bf16)vv.z; pk.h[3] = (__bf16)vv.w;
            *(uint2*)(bp + c0) = pk.u;
        }
    }
}

// ---------------------------------------------------------------------------
// Host launcher
// ---------------------------------------------------------------------------
extern "C" void kernel_launch(void* const* d_in, const int* in_sizes, int n_in,
                              void* d_out, int out_size, void* d_ws, size_t ws_size,
                              hipStream_t stream) {
    (void)in_sizes; (void)n_in; (void)out_size; (void)ws_size;

    const float* x    = (const float*)d_in[0];
    const float* ca   = (const float*)d_in[2];
    const float* Wq_s = (const float*)d_in[4];
    const float* Wk_s = (const float*)d_in[5];
    const float* Wv_s = (const float*)d_in[6];
    const float* Wo_s = (const float*)d_in[7];
    const float* bo_s = (const float*)d_in[8];
    const float* Wq_c = (const float*)d_in[9];
    const float* Wk_c = (const float*)d_in[10];
    const float* Wv_c = (const float*)d_in[11];
    const float* Wo_c = (const float*)d_in[12];
    const float* bo_c = (const float*)d_in[13];
    const float* ln1_g = (const float*)d_in[14];
    const float* ln1_b = (const float*)d_in[15];
    const float* ln2_g = (const float*)d_in[16];
    const float* ln2_b = (const float*)d_in[17];
    const float* ln3_g = (const float*)d_in[18];
    const float* ln3_b = (const float*)d_in[19];
    const float* W1   = (const float*)d_in[20];
    const float* b1   = (const float*)d_in[21];
    const float* W2   = (const float*)d_in[22];
    const float* b2   = (const float*)d_in[23];
    float* out = (float*)d_out;

    // workspace carve-up
    char* w = (char*)d_ws;
    auto alloc = [&](size_t bytes) {
        char* p = w; w += (bytes + 255) & ~(size_t)255; return p;
    };
    const size_t HED = (size_t)HH * EE * DD;
    __bf16* tWq_s = (__bf16*)alloc(HED * 2);               // [H, D, E]
    __bf16* tWk_s = (__bf16*)alloc(HED * 2);
    __bf16* tWv_s = (__bf16*)alloc(HED * 2);
    __bf16* tWo_s = (__bf16*)alloc((size_t)EE * EE * 2);   // [N=E, K=E]
    __bf16* tWq_c = (__bf16*)alloc(HED * 2);
    __bf16* tWk_c = (__bf16*)alloc(HED * 2);
    __bf16* tWv_c = (__bf16*)alloc(HED * 2);
    __bf16* tWo_c = (__bf16*)alloc((size_t)EE * EE * 2);
    __bf16* tW1   = (__bf16*)alloc((size_t)EE * E4 * 2);   // [4E, E]
    __bf16* tW2   = (__bf16*)alloc((size_t)E4 * EE * 2);   // [E, 4E]
    __bf16* xln   = (__bf16*)alloc((size_t)BT * EE * 2);
    __bf16* cab   = (__bf16*)alloc((size_t)BB * SS * EE * 2);
    __bf16* qb    = (__bf16*)alloc((size_t)BB * HH * TT * DD * 2);
    __bf16* kbuf  = (__bf16*)alloc((size_t)BB * HH * SS * DD * 2);
    __bf16* vt    = (__bf16*)alloc((size_t)BB * HH * SS * DD * 2); // [B,H,D,len]
    __bf16* aout  = (__bf16*)alloc((size_t)BT * EE * 2);
    float*  x1    = (float*)alloc((size_t)BT * EE * 4);
    __bf16* hid   = (__bf16*)alloc((size_t)BT * E4 * 2);

    auto cvtT = [&](const float* src, __bf16* dst, int batch, int R, int C) {
        dim3 g((unsigned)((R * C + 255) / 256), batch);
        cvtT_f32_bf16_kernel<<<g, 256, 0, stream>>>(src, dst, R, C);
    };
    cvtT(Wq_s, tWq_s, HH, EE, DD); cvtT(Wk_s, tWk_s, HH, EE, DD);
    cvtT(Wv_s, tWv_s, HH, EE, DD);
    cvtT(Wq_c, tWq_c, HH, EE, DD); cvtT(Wk_c, tWk_c, HH, EE, DD);
    cvtT(Wv_c, tWv_c, HH, EE, DD);
    cvtT(Wo_s, tWo_s, 1, EE, EE);  cvtT(Wo_c, tWo_c, 1, EE, EE);
    cvtT(W1,   tW1,   1, EE, E4);  cvtT(W2,   tW2,   1, E4, EE);
    cvt_f32_bf16_kernel<<<dim3((unsigned)(((size_t)BB * SS * EE + 255) / 256)),
                          256, 0, stream>>>(ca, cab, BB * SS * EE);

    const float scale = 0.03125f;  // E^-0.5 = 1/32 (reference scales by E)
    dim3 blk256(256), blk32(32);
    dim3 projGrid(BT / 256, HH);
    dim3 attnGrid(TT / 16, HH, BB);

    // ---- self attention ----
    layernorm_kernel<<<BT, blk256, 0, stream>>>(x, ln1_g, ln1_b, xln);
    headproj_kernel<<<projGrid, blk256, 0, stream>>>(xln, tWq_s, qb,   EE, 0);
    headproj_kernel<<<projGrid, blk256, 0, stream>>>(xln, tWk_s, kbuf, EE, 0);
    headproj_kernel<<<projGrid, blk256, 0, stream>>>(xln, tWv_s, vt,   EE, 1);
    attn_kernel<<<attnGrid, blk32, 0, stream>>>(qb, kbuf, vt, aout, TT, TT, 1, scale);
    gemm_bf16_kernel<<<dim3(BT / 128, EE / 128), blk256, 0, stream>>>(
        aout, tWo_s, bo_s, x, x1, nullptr, BT, EE, EE, 0);

    // ---- cross attention ----
    layernorm_kernel<<<BT, blk256, 0, stream>>>(x1, ln2_g, ln2_b, xln);
    headproj_kernel<<<projGrid, blk256, 0, stream>>>(xln, tWq_c, qb,   EE, 0);
    headproj_kernel<<<projGrid, blk256, 0, stream>>>(cab, tWk_c, kbuf, EE, 0);
    headproj_kernel<<<projGrid, blk256, 0, stream>>>(cab, tWv_c, vt,   EE, 1);
    attn_kernel<<<attnGrid, blk32, 0, stream>>>(qb, kbuf, vt, aout, TT, SS, 0, scale);
    gemm_bf16_kernel<<<dim3(BT / 128, EE / 128), blk256, 0, stream>>>(
        aout, tWo_c, bo_c, x1, out, nullptr, BT, EE, EE, 0);

    // ---- feed-forward ----
    layernorm_kernel<<<BT, blk256, 0, stream>>>(out, ln3_g, ln3_b, xln);
    gemm_bf16_kernel<<<dim3(BT / 128, E4 / 128), blk256, 0, stream>>>(
        xln, tW1, b1, nullptr, nullptr, hid, BT, E4, EE, 1);
    gemm_bf16_kernel<<<dim3(BT / 128, EE / 128), blk256, 0, stream>>>(
        hid, tW2, b2, out, out, nullptr, BT, EE, E4, 0);
}
